// W8A16LLqllm_40587440947566
// MI455X (gfx1250) — matrix operand order repaired and checked by reading, not compiled, not run on hardware
//
#include <hip/hip_runtime.h>
#include <stdint.h>

typedef __attribute__((ext_vector_type(16))) __bf16 v16bf;
typedef __attribute__((ext_vector_type(8)))  __bf16 v8bf;
typedef __attribute__((ext_vector_type(8)))  float  v8f;
typedef __attribute__((ext_vector_type(4)))  float  v4f;
typedef __attribute__((ext_vector_type(4)))  unsigned int v4u;
typedef __attribute__((ext_vector_type(8)))  int    v8i;
typedef __attribute__((ext_vector_type(4)))  int    v4i;

#define KDIM 4096
#define NDIM 4096
#define BM 128
#define BN 128
#define BK 32
#define NKSTEPS (KDIM / BK)

#if __has_builtin(__builtin_amdgcn_tensor_load_to_lds)
#define HAVE_TDM 1
#else
#define HAVE_TDM 0
#endif

// ---------------------------------------------------------------------------
// Kernel 1: one-shot dequant + transpose  W[N][K] int8  ->  Wt[K][N] bf16
// (int8 is exactly representable in bf16). 64x64 tiles through LDS so both
// the global reads and the global writes are coalesced.
// ---------------------------------------------------------------------------
__global__ __launch_bounds__(256) void dequant_transpose(
    const signed char* __restrict__ W, __bf16* __restrict__ Wt)
{
    __shared__ __align__(16) __bf16 t[64][72];   // stride 144B: keeps 16B align
    const int tid = threadIdx.x;
    const int n0  = blockIdx.x * 64;
    const int k0  = blockIdx.y * 64;

    {   // load 64n x 64k int8 tile, convert, store transposed into LDS
        const int n  = tid >> 2;          // 0..63
        const int kc = (tid & 3) * 16;    // 0,16,32,48
        const v4i raw = *(const v4i*)(W + (size_t)(n0 + n) * KDIM + (k0 + kc));
#pragma unroll
        for (int w = 0; w < 4; ++w) {
            const int word = raw[w];
#pragma unroll
            for (int b = 0; b < 4; ++b) {
                const int v = (word << (24 - 8 * b)) >> 24;   // sign-extend byte
                t[kc + w * 4 + b][n] = (__bf16)(float)v;
            }
        }
    }
    __syncthreads();
    {   // write 64k x 64n bf16 tile, rows contiguous in N
        const int k  = tid >> 2;          // 0..63
        const int nc = (tid & 3) * 16;    // 0,16,32,48
        const v8bf r0 = *(const v8bf*)&t[k][nc];
        const v8bf r1 = *(const v8bf*)&t[k][nc + 8];
        __bf16* dst = Wt + (size_t)(k0 + k) * NDIM + (n0 + nc);
        *(v8bf*)dst       = r0;
        *(v8bf*)(dst + 8) = r1;
    }
}

// ---------------------------------------------------------------------------
// TDM: issue one 2D tile DMA  Wt tile (BK rows x BN cols, bf16) -> LDS.
// D# packing per CDNA5 ISA ch.8 (group0: count/lds/global/type, group1:
// data_size, dims, tile dims, stride). 2D tensor -> groups 2/3 zero.
// ---------------------------------------------------------------------------
#if HAVE_TDM
__device__ __forceinline__ void tdm_load_b_tile(const __bf16* gsrc, unsigned lds_byte_addr)
{
    const uint64_t ga = (uint64_t)(uintptr_t)gsrc;
    v4u g0;
    g0[0] = 1u;                                            // count=1, user mode
    g0[1] = lds_byte_addr;                                 // lds_addr [63:32]
    g0[2] = (unsigned)(ga & 0xffffffffu);                  // global_addr lo
    g0[3] = (unsigned)((ga >> 32) & 0x01ffffffu) | (2u << 30); // addr hi | type=2
    v8i g1;
    g1[0] = (1 << 16);                                     // data_size=1 (2B), mask=0
    g1[1] = (int)((NDIM & 0xffff) << 16);                  // tensor_dim0 lo16 @bit48
    g1[2] = (int)(((NDIM >> 16) & 0xffff) | ((KDIM & 0xffff) << 16)); // dim0 hi|dim1 lo
    g1[3] = (int)(((KDIM >> 16) & 0xffff) | (BN << 16));   // dim1 hi | tile_dim0
    g1[4] = BK;                                            // tile_dim1 | tile_dim2=0
    g1[5] = NDIM;                                          // tensor_dim0_stride lo32
    g1[6] = 0;                                             // stride hi | dim1_stride lo
    g1[7] = 0;
    const v4i g2 = (v4i)0;
    const v4i g3 = (v4i)0;
#if __clang_major__ >= 23
    const v8i g4 = (v8i)0;
    __builtin_amdgcn_tensor_load_to_lds(g0, g1, g2, g3, g4, 0);
#else
    __builtin_amdgcn_tensor_load_to_lds(g0, g1, g2, g3, 0);
#endif
}
#endif

// ---------------------------------------------------------------------------
// Kernel 2: GEMM  out[M,N] = A[M,K] @ Wt[K,N] * scales + bias
// Block tile 128x128x32, 8 waves (2M x 4N), wave tile 64x32 -> 8 WMMA/K-step.
// Double-buffered LDS; B tile arrives via TDM (TENSORcnt) when available.
// PRE=true  : B from pre-dequantized Wt (TDM or vector copy)
// PRE=false : B dequantized inline from int8 W (no workspace fallback)
// ---------------------------------------------------------------------------
template <bool PRE>
__global__ __launch_bounds__(256) void w8a32_wmma_gemm(
    const float* __restrict__ A,        // [M, K] fp32
    const signed char* __restrict__ W,  // [N, K] int8
    const __bf16* __restrict__ Wt,      // [K, N] bf16 (PRE path)
    const float* __restrict__ scales,   // [N]
    const float* __restrict__ bias,     // [N]
    float* __restrict__ out,            // [M, N]
    int M)
{
    __shared__ __align__(16) __bf16 lA[2][BM * BK];   // [m][k], k contiguous
    __shared__ __align__(16) __bf16 lB[2][BK * BN];   // [k][n], n contiguous

    const int tid     = threadIdx.x;
    const int lane    = tid & 31;
    const int wave    = tid >> 5;
    const int waveM   = wave >> 2;     // 0..1 -> 64-row slice
    const int waveN   = wave & 3;      // 0..3 -> 32-col slice
    const int l15     = lane & 15;
    const int halfsel = lane >> 4;

    const int blockN0 = blockIdx.x * BN;
    const int blockM0 = blockIdx.y * BM;

#if HAVE_TDM
    const unsigned ldsB_base = (unsigned)(uintptr_t)&lB[0][0];
#endif

    v8f acc[4][2];
#pragma unroll
    for (int mt = 0; mt < 4; ++mt)
#pragma unroll
        for (int nt = 0; nt < 2; ++nt)
            acc[mt][nt] = (v8f)(0.0f);

    auto stageA = [&](int k0, int buf) {
#pragma unroll
        for (int c = 0; c < 2; ++c) {
            const int chunk = tid + c * 256;      // 0..511
            const int m     = chunk >> 2;         // 0..127
            const int kc    = (chunk & 3) * 8;    // 0,8,16,24
            const float* src = A + (size_t)(blockM0 + m) * KDIM + (k0 + kc);
            const v4f f0 = *(const v4f*)(src);
            const v4f f1 = *(const v4f*)(src + 4);
            v8bf h;
#pragma unroll
            for (int i = 0; i < 4; ++i) {
                h[i]     = (__bf16)f0[i];
                h[i + 4] = (__bf16)f1[i];
            }
            *(v8bf*)&lA[buf][m * BK + kc] = h;
        }
    };

    auto stageB = [&](int k0, int buf) {
        if constexpr (PRE) {
#if HAVE_TDM
            if (wave == 0)
                tdm_load_b_tile(Wt + (size_t)k0 * NDIM + blockN0,
                                ldsB_base + (unsigned)buf * (BK * BN * 2));
#else
            const int k    = tid >> 3;            // 0..31
            const int nseg = (tid & 7) * 16;      // 0..112
            const __bf16* src = Wt + (size_t)(k0 + k) * NDIM + blockN0 + nseg;
            *(v8bf*)&lB[buf][k * BN + nseg]     = *(const v8bf*)src;
            *(v8bf*)&lB[buf][k * BN + nseg + 8] = *(const v8bf*)(src + 8);
#endif
        } else {
            // inline int8 dequant + transpose (fallback, no workspace)
            const int n  = tid & 127;             // 0..127
            const int kc = (tid >> 7) * 16;       // 0 or 16
            const v4i raw = *(const v4i*)(W + (size_t)(blockN0 + n) * KDIM + (k0 + kc));
#pragma unroll
            for (int w = 0; w < 4; ++w) {
                const int word = raw[w];
#pragma unroll
                for (int b = 0; b < 4; ++b) {
                    const int v = (word << (24 - 8 * b)) >> 24;
                    lB[buf][(kc + w * 4 + b) * BN + n] = (__bf16)(float)v;
                }
            }
        }
    };

    auto waitB = [&]() {
#if HAVE_TDM
        if constexpr (PRE) {
            if (wave == 0) __builtin_amdgcn_s_wait_tensorcnt(0);
        }
#endif
    };

    // ---- pipeline prologue: fill buffer 0 ----
    stageB(0, 0);
    stageA(0, 0);
    waitB();
    __syncthreads();

    for (int i = 0; i < NKSTEPS; ++i) {
        const int p = i & 1;
        if (i + 1 < NKSTEPS) {              // stage next tile into 1-p
            stageB((i + 1) * BK, 1 - p);
            stageA((i + 1) * BK, 1 - p);
        }

        // ---- fragments from LDS buffer p (exact WMMA VGPR layouts) ----
        v16bf aF[4], bF[2];
#pragma unroll
        for (int mt = 0; mt < 4; ++mt) {
            const int m  = waveM * 64 + mt * 16 + l15;
            const int kb = halfsel * 8;
            union { v16bf v; v8bf h[2]; } u;
            u.h[0] = *(const v8bf*)&lA[p][m * BK + kb];
            u.h[1] = *(const v8bf*)&lA[p][m * BK + kb + 16];
            aF[mt] = u.v;
        }
#pragma unroll
        for (int nt = 0; nt < 2; ++nt) {
            const int n0 = waveN * 32 + nt * 16;
            v16bf b;
#pragma unroll
            for (int s = 0; s < 16; ++s)
                b[s] = lB[p][(halfsel * 8 + (s & 7) + (s >> 3) * 16) * BN + n0 + l15];
            bF[nt] = b;
        }

        // ---- 8 WMMAs per wave per K-step ----
#pragma unroll
        for (int mt = 0; mt < 4; ++mt)
#pragma unroll
            for (int nt = 0; nt < 2; ++nt)
                acc[mt][nt] = __builtin_amdgcn_wmma_f32_16x16x32_bf16(
                    false, aF[mt], false, bF[nt],
                    (short)0, acc[mt][nt], false, false);

        if (i + 1 < NKSTEPS) waitB();       // TDM for (i+1) done before barrier
        __syncthreads();
    }

    // ---- epilogue: per-channel scale + bias; lane owns a fixed n ----
#pragma unroll
    for (int nt = 0; nt < 2; ++nt) {
        const int n   = blockN0 + waveN * 32 + nt * 16 + l15;
        const float s  = scales[n];
        const float bi = bias[n];
#pragma unroll
        for (int mt = 0; mt < 4; ++mt) {
            const int mBase = blockM0 + waveM * 64 + mt * 16 + halfsel * 8;
#pragma unroll
            for (int j = 0; j < 8; ++j)
                out[(size_t)(mBase + j) * NDIM + n] = acc[mt][nt][j] * s + bi;
        }
    }
}

// ---------------------------------------------------------------------------
extern "C" void kernel_launch(void* const* d_in, const int* in_sizes, int n_in,
                              void* d_out, int out_size, void* d_ws, size_t ws_size,
                              hipStream_t stream) {
    (void)n_in; (void)out_size;
    const float*       A      = (const float*)d_in[0];
    const signed char* W      = (const signed char*)d_in[1];
    const float*       scales = (const float*)d_in[2];
    const float*       bias   = (const float*)d_in[3];
    float*             out    = (float*)d_out;

    const int M = in_sizes[0] / KDIM;               // B*S = 8192
    const dim3 grid(NDIM / BN, M / BM);
    const dim3 block(256);

    const size_t wtBytes = (size_t)KDIM * NDIM * 2; // bf16 W^T = 32 MB
    if (ws_size >= wtBytes) {
        __bf16* Wt = (__bf16*)d_ws;
        dequant_transpose<<<dim3(NDIM / 64, KDIM / 64), 256, 0, stream>>>(W, Wt);
        w8a32_wmma_gemm<true><<<grid, block, 0, stream>>>(A, W, Wt, scales, bias, out, M);
    } else {
        w8a32_wmma_gemm<false><<<grid, block, 0, stream>>>(A, W, nullptr, scales, bias, out, M);
    }
}
